// LogicalReasoningLayer_37864431681748
// MI455X (gfx1250) — matrix operand order repaired and checked by reading, not compiled, not run on hardware
//
#include <hip/hip_runtime.h>
#include <hip/hip_bf16.h>
#include <math.h>

typedef __attribute__((ext_vector_type(16))) _Float16 v16h;
typedef __attribute__((ext_vector_type(8)))  _Float16 v8h;
typedef __attribute__((ext_vector_type(8)))  float    v8f;

#define NM   8192      // B*S
#define HDIM 512
#define SEQ  2048
#define NOPS 6
#define NHEADS 4
#define HD   128

#if defined(__AMDGCN__) && __has_builtin(__builtin_amdgcn_global_load_async_to_lds_b128) && __has_builtin(__builtin_amdgcn_s_wait_asynccnt)
#define ASYNC_OK 1
#else
#define ASYNC_OK 0
#endif

__device__ __forceinline__ v8f wmma16(v16h a, v16h b, v8f c) {
  return __builtin_amdgcn_wmma_f32_16x16x32_f16(false, a, false, b, (short)0, c, false, false);
}

__device__ __forceinline__ float gelu_exact(float v) {
  return 0.5f * v * (1.0f + erff(v * 0.70710678118654752f));
}

__device__ __forceinline__ float wredsum(float v) {
#pragma unroll
  for (int o = 16; o > 0; o >>= 1) v += __shfl_xor(v, o);
  return v;
}

// 16 halves (32B) global -> LDS, async if available
__device__ __forceinline__ void ldasync16(_Float16* l, const _Float16* g) {
#if ASYNC_OK
  typedef int v4i_t __attribute__((__vector_size__(16)));
  typedef __attribute__((address_space(1))) v4i_t* gptr_t;
  typedef __attribute__((address_space(3))) v4i_t* lptr_t;
  __builtin_amdgcn_global_load_async_to_lds_b128(
      (gptr_t)(uintptr_t)g, (lptr_t)(uint32_t)(uintptr_t)l, 0, 0);
  __builtin_amdgcn_global_load_async_to_lds_b128(
      (gptr_t)(uintptr_t)(g + 8), (lptr_t)(uint32_t)((uintptr_t)l + 16), 0, 0);
#else
  *(v8h*)l = *(const v8h*)g;
  *(v8h*)(l + 8) = *(const v8h*)(g + 8);
#endif
}
__device__ __forceinline__ void wait_async4() {
#if ASYNC_OK
  __builtin_amdgcn_s_wait_asynccnt(4);
#endif
}
__device__ __forceinline__ void wait_async0() {
#if ASYNC_OK
  __builtin_amdgcn_s_wait_asynccnt(0);
#endif
}

// ---------------------------------------------------------------------------
// WMMA GEMM on f16 inputs:  O[m,n] = epilogue( sum_k A[m,k] * W[n,k] )
// 128 threads (4 waves), tile 64x64, K-step 32, double-buffered async LDS.
// ---------------------------------------------------------------------------
struct GemmP {
  const _Float16* A; int ldA;
  const _Float16* W; int ldW;
  float* O;       int ldO;       // optional f32 output
  _Float16* oh;   int ldOh;      // optional f16 output
  int K;
  const float* bias;   // per-n bias (may be null)
  int epi;             // 0=bias 1=bias+gelu 2=weighted-acc 3=qkv 4=rec-agg 5=gate
  const float* aux0;   // epi2: opw col (stride 6); epi5: g1
  const float* aux1;   // epi4: dvec;               epi5: x
  const float* aux3;   // epi5: enhanced
  float* aux2w;        // epi4: enhanced (accum)
  _Float16* oq; _Float16* ok; _Float16* ov;  // epi3 outputs
  float scale;         // epi4
};

#define LDSP 40

__global__ __launch_bounds__(128) void gemm_wmma(GemmP p) {
  __shared__ _Float16 As[2][64][LDSP];
  __shared__ _Float16 Bs[2][64][LDSP];
  const int tid  = threadIdx.x;
  const int lane = tid & 31;
  const int wave = tid >> 5;
  const int wm = (wave >> 1) * 32;
  const int wn = (wave & 1) * 32;
  const int m0 = blockIdx.y * 64;
  const int n0 = blockIdx.x * 64;

  const int lr = tid >> 1;          // 0..63 row staged by this thread
  const int lc = (tid & 1) << 4;    // half-col 0 / 16

  v8f acc[2][2] = {};

  const _Float16* agBase = p.A + (size_t)(m0 + lr) * p.ldA + lc;
  const _Float16* wgBase = p.W + (size_t)(n0 + lr) * p.ldW + lc;

  ldasync16(&As[0][lr][lc], agBase);
  ldasync16(&Bs[0][lr][lc], wgBase);

  const int nsteps = p.K >> 5;
  for (int s = 0; s < nsteps; s++) {
    const int buf = s & 1;
    if (s + 1 < nsteps) {
      const int k1 = (s + 1) << 5;
      ldasync16(&As[1 - buf][lr][lc], agBase + k1);
      ldasync16(&Bs[1 - buf][lr][lc], wgBase + k1);
      wait_async4();          // stage s complete (in-order)
    } else {
      wait_async0();
    }
    __syncthreads();

    const int ar  = lane & 15;
    const int kb  = (lane >> 4) << 3;   // A frag: halves 0-7 = K kb..+7, 8-15 = K kb+16..+23
    const int kbb = (lane >> 4) << 3;   // B frag: halves 0-7 = K kbb..+7, 8-15 = K kbb+16..+23
    v16h a[2], b[2];
#pragma unroll
    for (int t = 0; t < 2; t++) {
      const _Float16* pr = &As[buf][wm + t * 16 + ar][0];
      v8h lo = *(const v8h*)(pr + kb);
      v8h hi = *(const v8h*)(pr + kb + 16);
#pragma unroll
      for (int i = 0; i < 8; i++) { a[t][i] = lo[i]; a[t][i + 8] = hi[i]; }
      const _Float16* pc = &Bs[buf][wn + t * 16 + ar][0];
      v8h bl = *(const v8h*)(pc + kbb);
      v8h bh = *(const v8h*)(pc + kbb + 16);
#pragma unroll
      for (int i = 0; i < 8; i++) { b[t][i] = bl[i]; b[t][i + 8] = bh[i]; }
    }
#pragma unroll
    for (int i = 0; i < 2; i++)
#pragma unroll
      for (int j = 0; j < 2; j++)
        acc[i][j] = wmma16(a[i], b[j], acc[i][j]);
    __syncthreads();
  }

  // C layout: lane holds col n = lane&15, rows m = (lane>>4)*8 + r
  const int cn = lane & 15;
  const int rg = (lane >> 4) << 3;
#pragma unroll
  for (int si = 0; si < 2; si++)
#pragma unroll
    for (int sj = 0; sj < 2; sj++) {
      const int gnn = n0 + wn + sj * 16 + cn;
      const float bz = p.bias ? p.bias[gnn] : 0.0f;
#pragma unroll
      for (int r = 0; r < 8; r++) {
        const int gm = m0 + wm + si * 16 + rg + r;
        float v = acc[si][sj][r];
        switch (p.epi) {
          case 0: {
            float z = v + bz;
            if (p.O)  p.O[(size_t)gm * p.ldO + gnn] = z;
            if (p.oh) p.oh[(size_t)gm * p.ldOh + gnn] = (_Float16)z;
          } break;
          case 1: {
            float z = gelu_exact(v + bz);
            if (p.O)  p.O[(size_t)gm * p.ldO + gnn] = z;
            if (p.oh) p.oh[(size_t)gm * p.ldOh + gnn] = (_Float16)z;
          } break;
          case 2: { // enhanced += opw[m,o] * (acc + bias)
            p.O[(size_t)gm * p.ldO + gnn] += p.aux0[(size_t)gm * NOPS] * (v + bz);
          } break;
          case 3: { // qkv split, f16 outputs; V transposed to [B,NH,HD,S]
            float z = v + bz;
            if (gnn < HDIM)            p.oq[(size_t)gm * HDIM + gnn] = (_Float16)z;
            else if (gnn < 2 * HDIM)   p.ok[(size_t)gm * HDIM + (gnn - HDIM)] = (_Float16)z;
            else {
              int c = gnn - 2 * HDIM;
              int hh = c >> 7, dd = c & 127;
              int bb = gm >> 11, ss = gm & 2047;
              p.ov[(((size_t)(bb * NHEADS + hh)) * HD + dd) * SEQ + ss] = (_Float16)z;
            }
          } break;
          case 4: { // rec(f16) = acc + dvec[n]; enhanced(f32) += rec * scale
            float z = v + p.aux1[gnn];
            if (p.oh) p.oh[(size_t)gm * p.ldOh + gnn] = (_Float16)z;
            p.aux2w[(size_t)gm * HDIM + gnn] += z * p.scale;
          } break;
          case 5: { // gate blend
            float z  = v + p.aux0[(size_t)gm * HDIM + gnn] + bz;
            float g  = 1.0f / (1.0f + __expf(-z));
            float xv = p.aux1[(size_t)gm * HDIM + gnn];
            float ev = p.aux3[(size_t)gm * HDIM + gnn];
            p.O[(size_t)gm * p.ldO + gnn] = xv + g * (ev - xv);
          } break;
        }
      }
    }
}

// ---------------------------------------------------------------------------
// Flash attention, one wave per (b, head, 16-query tile).
// S^T = K @ Q^T (softmax over keys is in-lane + xor16); ctx^T = V^T @ P^T.
// ---------------------------------------------------------------------------
__global__ __launch_bounds__(256) void attn_flash(const _Float16* __restrict__ Qh,
                                                  const _Float16* __restrict__ Kh,
                                                  const _Float16* __restrict__ Vt,
                                                  _Float16* __restrict__ ctx) {
  const int lane = threadIdx.x & 31;
  const int wid  = blockIdx.x * 8 + (threadIdx.x >> 5);
  const int qt = wid & 127;
  const int hh = (wid >> 7) & 3;
  const int bb = wid >> 9;
  const int qrow0 = bb * SEQ + qt * 16;
  const int hcol  = hh * HD;
  const float sc = 0.088388347648318447f;  // 1/sqrt(128)

  v16h bq[4];
  {
    const _Float16* qp = Qh + (size_t)(qrow0 + (lane & 15)) * HDIM + hcol + ((lane >> 4) << 3);
#pragma unroll
    for (int c = 0; c < 4; c++) {
      v8h lo = *(const v8h*)(qp + c * 32);
      v8h hi = *(const v8h*)(qp + c * 32 + 16);
#pragma unroll
      for (int i = 0; i < 8; i++) { bq[c][i] = lo[i]; bq[c][i + 8] = hi[i]; }
    }
  }

  v8f oc[8] = {};
  float rm = -3.0e38f, rl = 0.0f;
  const int kb = (lane >> 4) << 3;
  const size_t vbase = ((size_t)(bb * NHEADS + hh) * HD) * SEQ;

  for (int kw = 0; kw < SEQ; kw += 32) {
    v8f s0 = {}, s1 = {};
#pragma unroll
    for (int c = 0; c < 4; c++) {
      const _Float16* kp = Kh + (size_t)(bb * SEQ + kw + (lane & 15)) * HDIM + hcol + c * 32 + kb;
      v8h lo = *(const v8h*)kp;
      v8h hi = *(const v8h*)(kp + 16);
      v16h a;
#pragma unroll
      for (int i = 0; i < 8; i++) { a[i] = lo[i]; a[i + 8] = hi[i]; }
      s0 = wmma16(a, bq[c], s0);
      const _Float16* kp1 = kp + (size_t)16 * HDIM;
      lo = *(const v8h*)kp1;
      hi = *(const v8h*)(kp1 + 16);
#pragma unroll
      for (int i = 0; i < 8; i++) { a[i] = lo[i]; a[i + 8] = hi[i]; }
      s1 = wmma16(a, bq[c], s1);
    }
    float wmax = -3.0e38f;
#pragma unroll
    for (int i = 0; i < 8; i++) {
      s0[i] *= sc; s1[i] *= sc;
      wmax = fmaxf(wmax, fmaxf(s0[i], s1[i]));
    }
    wmax = fmaxf(wmax, __shfl_xor(wmax, 16));
    float mnew = fmaxf(rm, wmax);
    float cf = __expf(rm - mnew);
    float wsum = 0.0f;
#pragma unroll
    for (int i = 0; i < 8; i++) {
      s0[i] = __expf(s0[i] - mnew);
      s1[i] = __expf(s1[i] - mnew);
      wsum += s0[i] + s1[i];
    }
    wsum += __shfl_xor(wsum, 16);
    rl = rl * cf + wsum;
    rm = mnew;
#pragma unroll
    for (int dt = 0; dt < 8; dt++)
#pragma unroll
      for (int i = 0; i < 8; i++) oc[dt][i] *= cf;

    v16h bp;
    const bool lowg = lane < 16;
#pragma unroll
    for (int i = 0; i < 8; i++) {
      float x0 = __shfl_xor(s0[i], 16);
      float x1 = __shfl_xor(s1[i], 16);
      bp[i]     = (_Float16)(lowg ? s0[i] : x1);
      bp[i + 8] = (_Float16)(lowg ? x0 : s1[i]);
    }
#pragma unroll
    for (int dt = 0; dt < 8; dt++) {
      const _Float16* vp = Vt + vbase + (size_t)(dt * 16 + (lane & 15)) * SEQ + kw + ((lane >> 4) << 4);
      v8h lo = *(const v8h*)vp;
      v8h hi = *(const v8h*)(vp + 8);
      v16h a;
#pragma unroll
      for (int i = 0; i < 8; i++) { a[i] = lo[i]; a[i + 8] = hi[i]; }
      oc[dt] = wmma16(a, bp, oc[dt]);
    }
  }

  const float inv = 1.0f / rl;
  _Float16* op = ctx + (size_t)(qrow0 + (lane & 15)) * HDIM + hcol;
  const int rg = (lane >> 4) << 3;
#pragma unroll
  for (int dt = 0; dt < 8; dt++)
#pragma unroll
    for (int r = 0; r < 8; r++)
      op[dt * 16 + rg + r] = (_Float16)(oc[dt][r] * inv);
}

// ---------------------------------------------------------------------------
// Small VALU kernels
// ---------------------------------------------------------------------------
__global__ __launch_bounds__(256) void k_copy(const float* a, float* o, size_t n) {
  size_t i = (size_t)blockIdx.x * 256 + threadIdx.x;
  if (i < n) o[i] = a[i];
}

__global__ __launch_bounds__(256) void k_cvt(const float* a, _Float16* o, size_t n) {
  size_t i = (size_t)blockIdx.x * 256 + threadIdx.x;
  if (i < n) o[i] = (_Float16)a[i];
}

// weight rows (length 512) -> f16 pool: dst[r*512+c] = src[r*ldsrc + off + c]
__global__ __launch_bounds__(256) void k_wcvt(const float* __restrict__ src, int ldsrc, int off,
                                              _Float16* __restrict__ dst) {
  int i = blockIdx.x * 256 + threadIdx.x;
  int r = i >> 9, c = i & 511;
  dst[i] = (_Float16)src[(size_t)r * ldsrc + off + c];
}

__global__ __launch_bounds__(256) void k_opsel(const float* __restrict__ t0,
                                               const float* __restrict__ W2,
                                               const float* __restrict__ b2,
                                               float* __restrict__ opw) {
  const int m = blockIdx.x * 8 + (threadIdx.x >> 5);
  const int lane = threadIdx.x & 31;
  const float* xr = t0 + (size_t)m * HDIM;
  float s[NOPS] = {};
  for (int i = lane; i < HDIM; i += 32) {
    float xv = xr[i];
#pragma unroll
    for (int j = 0; j < NOPS; j++) s[j] += xv * W2[j * HDIM + i];
  }
#pragma unroll
  for (int j = 0; j < NOPS; j++) s[j] = wredsum(s[j]) + b2[j];
  float mx = s[0];
#pragma unroll
  for (int j = 1; j < NOPS; j++) mx = fmaxf(mx, s[j]);
  float den = 0.0f;
#pragma unroll
  for (int j = 0; j < NOPS; j++) { s[j] = __expf(s[j] - mx); den += s[j]; }
  if (lane < NOPS) opw[(size_t)m * NOPS + lane] = s[lane] / den;
}

__global__ __launch_bounds__(256) void k_c1(const float* __restrict__ opW1,
                                            const float* __restrict__ op_emb,
                                            const float* __restrict__ op_b1,
                                            float* __restrict__ c1) {
  const int n = blockIdx.x * 8 + (threadIdx.x >> 5);
  const int lane = threadIdx.x & 31;
  const int o = n >> 9;
  const float* wr = opW1 + (size_t)n * 1024 + 512;
  const float* e = op_emb + o * HDIM;
  float s = 0.0f;
  for (int i = lane; i < HDIM; i += 32) s += e[i] * wr[i];
  s = wredsum(s);
  if (lane == 0) c1[n] = s + op_b1[n];
}

// LN over k then gelu: f32 in -> f16 out (same [m,o,k] layout)
__global__ __launch_bounds__(256) void k_lngelu(const float* __restrict__ pre,
                                                _Float16* __restrict__ out,
                                                const float* __restrict__ g,
                                                const float* __restrict__ b) {
  const int rid = blockIdx.x * 8 + (threadIdx.x >> 5);
  const int lane = threadIdx.x & 31;
  const int o = rid % NOPS;
  const float* row = pre + (size_t)rid * HDIM;
  _Float16* orow = out + (size_t)rid * HDIM;
  const float* gg = g + o * HDIM;
  const float* bb = b + o * HDIM;
  float vals[16], s = 0.0f;
#pragma unroll
  for (int t = 0; t < 16; t++) { vals[t] = row[lane + 32 * t]; s += vals[t]; }
  s = wredsum(s);
  const float mean = s * (1.0f / HDIM);
  float s2 = 0.0f;
#pragma unroll
  for (int t = 0; t < 16; t++) { float d = vals[t] - mean; s2 += d * d; }
  s2 = wredsum(s2);
  const float rstd = rsqrtf(s2 * (1.0f / HDIM) + 1e-5f);
#pragma unroll
  for (int t = 0; t < 16; t++) {
    int idx = lane + 32 * t;
    orow[idx] = (_Float16)gelu_exact((vals[t] - mean) * rstd * gg[idx] + bb[idx]);
  }
}

__global__ __launch_bounds__(256) void k_dvec(const float* __restrict__ dep,
                                              const float* __restrict__ agg_w,
                                              const float* __restrict__ agg_b,
                                              float* __restrict__ dvec) {
  const int n = blockIdx.x * 8 + (threadIdx.x >> 5);
  const int lane = threadIdx.x & 31;
  const float* wr = agg_w + (size_t)n * 1024 + 512;
  float s = 0.0f;
  for (int i = lane; i < HDIM; i += 32) s += dep[i] * wr[i];
  s = wredsum(s);
  if (lane == 0) dvec[n] = s + agg_b[n];
}

// LayerNorm rows of 512: f32 in -> f16 out
__global__ __launch_bounds__(256) void k_ln(const float* __restrict__ in, _Float16* __restrict__ out,
                                            const float* __restrict__ g, const float* __restrict__ b) {
  const int row = blockIdx.x * 8 + (threadIdx.x >> 5);
  const int lane = threadIdx.x & 31;
  const float* r = in + (size_t)row * HDIM;
  float vals[16], s = 0.0f;
#pragma unroll
  for (int t = 0; t < 16; t++) { vals[t] = r[lane + 32 * t]; s += vals[t]; }
  s = wredsum(s);
  const float mean = s * (1.0f / HDIM);
  float s2 = 0.0f;
#pragma unroll
  for (int t = 0; t < 16; t++) { float d = vals[t] - mean; s2 += d * d; }
  s2 = wredsum(s2);
  const float rstd = rsqrtf(s2 * (1.0f / HDIM) + 1e-5f);
  _Float16* orow = out + (size_t)row * HDIM;
#pragma unroll
  for (int t = 0; t < 16; t++) {
    int idx = lane + 32 * t;
    orow[idx] = (_Float16)((vals[t] - mean) * rstd * g[idx] + b[idx]);
  }
}

// ---------------------------------------------------------------------------
extern "C" void kernel_launch(void* const* d_in, const int* in_sizes, int n_in,
                              void* d_out, int out_size, void* d_ws, size_t ws_size,
                              hipStream_t stream) {
  (void)in_sizes; (void)n_in; (void)out_size; (void)ws_size;
  const float* x          = (const float*)d_in[0];
  const float* op_emb     = (const float*)d_in[1];
  const float* op_W1      = (const float*)d_in[2];
  const float* op_b1      = (const float*)d_in[3];
  const float* op_ln_g    = (const float*)d_in[4];
  const float* op_ln_b    = (const float*)d_in[5];
  const float* op_W2      = (const float*)d_in[6];
  const float* op_b2      = (const float*)d_in[7];
  const float* sel_W1     = (const float*)d_in[8];
  const float* sel_b1     = (const float*)d_in[9];
  const float* sel_W2     = (const float*)d_in[10];
  const float* sel_b2     = (const float*)d_in[11];
  const float* attn_in_w  = (const float*)d_in[12];
  const float* attn_in_b  = (const float*)d_in[13];
  const float* attn_out_w = (const float*)d_in[14];
  const float* attn_out_b = (const float*)d_in[15];
  const float* rec_proj_w = (const float*)d_in[16];
  const float* rec_proj_b = (const float*)d_in[17];
  const float* rec_agg_w  = (const float*)d_in[18];
  const float* rec_agg_b  = (const float*)d_in[19];
  const float* depth_emb  = (const float*)d_in[20];
  const float* gate_w     = (const float*)d_in[21];
  const float* gate_b     = (const float*)d_in[22];
  const float* out_ln_g   = (const float*)d_in[23];
  const float* out_ln_b   = (const float*)d_in[24];
  const float* out_w      = (const float*)d_in[25];
  const float* out_b      = (const float*)d_in[26];

  char* wp = (char*)d_ws;
  auto alloc = [&](size_t bytes) -> void* {
    void* p = wp; wp += (bytes + 255) & ~(size_t)255; return p;
  };
  // f32 buffers
  float* t0   = (float*)alloc((size_t)NM * HDIM * 4);
  float* opw  = (float*)alloc((size_t)NM * NOPS * 4);
  float* c1   = (float*)alloc(3072 * 4);
  float* pre  = (float*)alloc((size_t)NM * 3072 * 4);
  float* enh  = (float*)alloc((size_t)NM * HDIM * 4);
  float* g1   = (float*)alloc((size_t)NM * HDIM * 4);
  float* fin  = (float*)alloc((size_t)NM * HDIM * 4);
  float* dvec = (float*)alloc(HDIM * 4);
  // f16 activations
  _Float16* xh    = (_Float16*)alloc((size_t)NM * HDIM * 2);
  _Float16* preh  = (_Float16*)alloc((size_t)NM * 3072 * 2);
  _Float16* projh = (_Float16*)alloc((size_t)NM * HDIM * 2);
  _Float16* ctxh  = (_Float16*)alloc((size_t)NM * HDIM * 2);
  _Float16* atth  = (_Float16*)alloc((size_t)NM * HDIM * 2);
  _Float16* rech  = (_Float16*)alloc((size_t)NM * HDIM * 2);
  _Float16* enhh  = (_Float16*)alloc((size_t)NM * HDIM * 2);
  _Float16* nrmh  = (_Float16*)alloc((size_t)NM * HDIM * 2);
  _Float16* Qh    = (_Float16*)alloc((size_t)NM * HDIM * 2);
  _Float16* Kh    = (_Float16*)alloc((size_t)NM * HDIM * 2);
  _Float16* Vt    = (_Float16*)alloc((size_t)NM * HDIM * 2);
  // f16 weight pool (all rows length 512)
  _Float16* selW1h   = (_Float16*)alloc((size_t)512 * 512 * 2);
  _Float16* opW1h    = (_Float16*)alloc((size_t)3072 * 512 * 2);
  _Float16* opW2h    = (_Float16*)alloc((size_t)3072 * 512 * 2);
  _Float16* attnInh  = (_Float16*)alloc((size_t)1536 * 512 * 2);
  _Float16* attnOuth = (_Float16*)alloc((size_t)512 * 512 * 2);
  _Float16* recProjh = (_Float16*)alloc((size_t)512 * 512 * 2);
  _Float16* recAggAh = (_Float16*)alloc((size_t)512 * 512 * 2);
  _Float16* gateAh   = (_Float16*)alloc((size_t)512 * 512 * 2);
  _Float16* gateBh   = (_Float16*)alloc((size_t)512 * 512 * 2);
  _Float16* outWh    = (_Float16*)alloc((size_t)512 * 512 * 2);

  auto wcvt = [&](const float* src, int ld, int off, _Float16* dst, int rows) {
    k_wcvt<<<(rows * 512) / 256, 256, 0, stream>>>(src, ld, off, dst);
  };
  auto gemm = [&](GemmP p, int M, int N) {
    gemm_wmma<<<dim3(N / 64, M / 64), dim3(128), 0, stream>>>(p);
  };

  // 0) one-time f16 conversions
  wcvt(sel_W1, 512, 0, selW1h, 512);
  wcvt(op_W1, 1024, 0, opW1h, 3072);
  wcvt(op_W2, 512, 0, opW2h, 3072);
  wcvt(attn_in_w, 512, 0, attnInh, 1536);
  wcvt(attn_out_w, 512, 0, attnOuth, 512);
  wcvt(rec_proj_w, 512, 0, recProjh, 512);
  wcvt(rec_agg_w, 1024, 0, recAggAh, 512);
  wcvt(gate_w, 1024, 0, gateAh, 512);
  wcvt(gate_w, 1024, 512, gateBh, 512);
  wcvt(out_w, 512, 0, outWh, 512);
  k_cvt<<<(NM * HDIM) / 256, 256, 0, stream>>>(x, xh, (size_t)NM * HDIM);

  // 1) t0 = gelu(x @ sel_W1.T + sel_b1)
  { GemmP p{}; p.A = xh; p.ldA = HDIM; p.W = selW1h; p.ldW = HDIM; p.O = t0; p.ldO = HDIM;
    p.K = HDIM; p.bias = sel_b1; p.epi = 1; gemm(p, NM, HDIM); }
  // 2) op_weights = softmax(t0 @ sel_W2.T + b2)
  k_opsel<<<NM / 8, 256, 0, stream>>>(t0, sel_W2, sel_b2, opw);
  // 3) c1 bias for pre
  k_c1<<<3072 / 8, 256, 0, stream>>>(op_W1, op_emb, op_b1, c1);
  // 4) pre = x @ W1x.T + c1
  { GemmP p{}; p.A = xh; p.ldA = HDIM; p.W = opW1h; p.ldW = HDIM; p.O = pre; p.ldO = 3072;
    p.K = HDIM; p.bias = c1; p.epi = 0; gemm(p, NM, 3072); }
  // 5) h = gelu(ln(pre)) -> preh (f16)
  k_lngelu<<<(NM * NOPS) / 8, 256, 0, stream>>>(pre, preh, op_ln_g, op_ln_b);
  // 6) enhanced = x
  k_copy<<<(NM * HDIM) / 256, 256, 0, stream>>>(x, enh, (size_t)NM * HDIM);
  // 7) enhanced += opw[:,o] * (h[:,o,:] @ op_W2[o].T + op_b2[o])
  for (int o = 0; o < NOPS; o++) {
    GemmP p{}; p.A = preh + o * HDIM; p.ldA = 3072;
    p.W = opW2h + (size_t)o * HDIM * HDIM; p.ldW = HDIM;
    p.O = enh; p.ldO = HDIM; p.K = HDIM; p.bias = op_b2 + o * HDIM;
    p.epi = 2; p.aux0 = opw + o; gemm(p, NM, HDIM);
  }
  // 8) rec = enhanced (f16 copy for GEMM A)
  k_cvt<<<(NM * HDIM) / 256, 256, 0, stream>>>(enh, rech, (size_t)NM * HDIM);

  for (int d = 0; d < 3; d++) {
    // proj = rec @ rec_proj_w.T + b  -> f16 only
    { GemmP p{}; p.A = rech; p.ldA = HDIM; p.W = recProjh; p.ldW = HDIM; p.oh = projh;
      p.ldOh = HDIM; p.K = HDIM; p.bias = rec_proj_b; p.epi = 0; gemm(p, NM, HDIM); }
    // qkv = proj @ attn_in_w.T + b -> Qh, Kh, Vt (f16, V transposed)
    { GemmP p{}; p.A = projh; p.ldA = HDIM; p.W = attnInh; p.ldW = HDIM;
      p.K = HDIM; p.bias = attn_in_b; p.epi = 3;
      p.oq = Qh; p.ok = Kh; p.ov = Vt; gemm(p, NM, 3 * HDIM); }
    // flash attention -> ctxh (f16)
    attn_flash<<<(4 * NHEADS * (SEQ / 16)) / 8, 256, 0, stream>>>(Qh, Kh, Vt, ctxh);
    // att = ctx @ attn_out_w.T + b -> f16 only
    { GemmP p{}; p.A = ctxh; p.ldA = HDIM; p.W = attnOuth; p.ldW = HDIM; p.oh = atth;
      p.ldOh = HDIM; p.K = HDIM; p.bias = attn_out_b; p.epi = 0; gemm(p, NM, HDIM); }
    // dvec = depth_emb[d] @ Wa_dep.T + rec_agg_b
    k_dvec<<<HDIM / 8, 256, 0, stream>>>(depth_emb + d * HDIM, rec_agg_w, rec_agg_b, dvec);
    // rec = att @ Wa_att.T + dvec ; enhanced += rec * 2^-(d+1)
    { GemmP p{}; p.A = atth; p.ldA = HDIM; p.W = recAggAh; p.ldW = HDIM; p.oh = rech;
      p.ldOh = HDIM; p.K = HDIM; p.bias = nullptr; p.epi = 4; p.aux1 = dvec; p.aux2w = enh;
      p.scale = 1.0f / (float)(1 << (d + 1)); gemm(p, NM, HDIM); }
  }

  // gate
  { GemmP p{}; p.A = xh; p.ldA = HDIM; p.W = gateAh; p.ldW = HDIM; p.O = g1; p.ldO = HDIM;
    p.K = HDIM; p.bias = nullptr; p.epi = 0; gemm(p, NM, HDIM); }
  k_cvt<<<(NM * HDIM) / 256, 256, 0, stream>>>(enh, enhh, (size_t)NM * HDIM);
  { GemmP p{}; p.A = enhh; p.ldA = HDIM; p.W = gateBh; p.ldW = HDIM; p.O = fin; p.ldO = HDIM;
    p.K = HDIM; p.bias = gate_b; p.epi = 5;
    p.aux0 = g1; p.aux1 = x; p.aux3 = enh; gemm(p, NM, HDIM); }

  // output LN + final projection
  k_ln<<<NM / 8, 256, 0, stream>>>(fin, nrmh, out_ln_g, out_ln_b);
  { GemmP p{}; p.A = nrmh; p.ldA = HDIM; p.W = outWh; p.ldW = HDIM; p.O = (float*)d_out;
    p.ldO = HDIM; p.K = HDIM; p.bias = out_b; p.epi = 0; gemm(p, NM, HDIM); }
}
